// TensorEmbedding_51110110822522
// MI455X (gfx1250) — compile-verified, matrix operand order checked by reading, not run on hardware
//
#include <hip/hip_runtime.h>
#include <math.h>
#include <stdint.h>

#define DEVI __device__ __forceinline__

typedef __attribute__((ext_vector_type(16))) _Float16 v16h;
typedef __attribute__((ext_vector_type(8)))  _Float16 v8h;
typedef __attribute__((ext_vector_type(8)))  float    v8f;
typedef __attribute__((ext_vector_type(4)))  unsigned int u32x4;
typedef __attribute__((ext_vector_type(8)))  int      i32x8;

static constexpr int N_NODES = 16384;
static constexpr int N_EDGES = 262144;

// ---------------- workspace layout (bytes) ----------------
static constexpr size_t OFF_X16   = 0;                    // 16384*64 f16      = 2,097,152
static constexpr size_t OFF_DPF   = 2097152;              // 3*64*32 f16       =    12,288
static constexpr size_t OFF_EMB2F = 2109440;              // 64*128 f16        =    16,384  (contiguous after DPF)
static constexpr size_t OFF_LS0F  = 2125824;              // 128*64 f16        =    16,384
static constexpr size_t OFF_LS1F  = 2142208;              // 192*128 f16       =    49,152
static constexpr size_t OFF_LTF   = 2191360;              // 3*64*64 f16       =    24,576
static constexpr size_t OFF_ACC   = 2215936;              // 16384*640 f32     = 41,943,040
static constexpr size_t OFF_NRM   = 44158976;             // 16384*64 f32      =  4,194,304
static constexpr size_t OFF_NFAC  = 48353280;             // 16384*192 f32     = 12,582,912

// edge-kernel LDS staging: dp (3*2048 f16) ++ emb2 (64*128 f16) = 14336 f16 = 28672 B
static constexpr int WT_F16   = 14336;
static constexpr int WT_BYTES = 28672;
static constexpr int EMB2_OFS = 6144;   // f16 index where emb2 block starts

// ---------------- WMMA helpers ----------------
DEVI v8f wmma(v16h a, v16h b, v8f c) {
  return __builtin_amdgcn_wmma_f32_16x16x32_f16(false, a, false, b, (short)0, c,
                                                false, false);
}

// A/B fragment, 16x32 (M/N x K), K pattern: lanes<16 -> K {0..7,16..23};
// lanes>=16 -> K {8..15,24..31}.  `row` points at the 32-long K row.
DEVI v16h frag_f32(const float* row, int hi) {
  v16h a;
  const float* p0 = row + hi * 8;
  const float* p1 = row + 16 + hi * 8;
#pragma unroll
  for (int i = 0; i < 8; ++i) { a[i] = (_Float16)p0[i]; a[8 + i] = (_Float16)p1[i]; }
  return a;
}
DEVI v16h frag_f16(const _Float16* row, int hi) {
  v8h lo = *(const v8h*)(row + hi * 8);
  v8h hh = *(const v8h*)(row + 16 + hi * 8);
  v16h a;
#pragma unroll
  for (int i = 0; i < 8; ++i) { a[i] = lo[i]; a[8 + i] = hh[i]; }
  return a;
}

DEVI float silu(float x) { return x / (1.0f + __expf(-x)); }

// ---------------- prep kernels ----------------
__global__ void k_cvt_f16(const float* __restrict__ src, _Float16* __restrict__ dst, int n) {
  int i = blockIdx.x * blockDim.x + threadIdx.x;
  if (i < n) dst[i] = (_Float16)src[i];
}

__global__ void k_gather_x(const int* __restrict__ z, const float* __restrict__ emb_w,
                           _Float16* __restrict__ x16) {
  int i = blockIdx.x * blockDim.x + threadIdx.x; // n*64+u
  int n = i >> 6, u = i & 63;
  x16[i] = (_Float16)emb_w[z[n] * 64 + u];
}

__global__ void k_zero(float* __restrict__ p, int n) {
  int i = blockIdx.x * blockDim.x + threadIdx.x;
  if (i < n) p[i] = 0.0f;
}

// ---------------- edge kernel: GEMMs + fused scatter ----------------
// block = 256 (8 waves), wave handles 16 edges; grid = N_EDGES/128 = 2048
__global__ void k_edges(const int* __restrict__ eidx, const float* __restrict__ ew,
                        const float* __restrict__ evec, const float* __restrict__ eattr,
                        const float* __restrict__ dp1b, const float* __restrict__ dp2b,
                        const float* __restrict__ dp3b, const float* __restrict__ emb2b,
                        const _Float16* __restrict__ x16,
                        const _Float16* __restrict__ wsrc,  // dpf ++ emb2f, 28672 B
                        float* __restrict__ acc) {
  const int wave = threadIdx.x >> 5, lane = threadIdx.x & 31;
  const int hi = lane >> 4, lm = lane & 15;
  const int ebase = (blockIdx.x * 8 + wave) * 16;

  __shared__ float s_vx[8][16], s_vy[8][16], s_vz[8][16], s_C[8][16];
  __shared__ int   s_src[8][16], s_dst[8][16];
  __shared__ __align__(16) _Float16 s_wt[WT_F16];

  // --- stage shared B-matrices into LDS via the Tensor Data Mover -------------
  if (threadIdx.x == 0) {
    uint64_t ga  = (uint64_t)(uintptr_t)wsrc;
    uint32_t lds = (uint32_t)(uintptr_t)&s_wt[0];
    const unsigned NE = WT_BYTES / 8;               // 3584 x 8-byte elements, 1-D tile
    u32x4 g0;
    g0[0] = 1u;                                      // count=1, user descriptor
    g0[1] = lds;                                     // lds_addr (bytes)
    g0[2] = (unsigned)ga;                            // global_addr[31:0]
    g0[3] = (unsigned)((ga >> 32) & 0x01FFFFFFu) | (2u << 30);  // addr[56:32] | type=2
    i32x8 g1;
    g1[0] = (int)(3u << 16);                         // data_size = 8 B
    g1[1] = (int)((NE & 0xFFFFu) << 16);             // tensor_dim0[15:0]
    g1[2] = (int)((NE >> 16) | (1u << 16));          // tensor_dim0[31:16] | tensor_dim1=1
    g1[3] = (int)((NE & 0xFFFFu) << 16);             // tile_dim0 = NE (tensor_dim1 hi = 0)
    g1[4] = 1;                                       // tile_dim1 = 1, tile_dim2 = 0
    g1[5] = (int)NE;                                 // tensor_dim0_stride[31:0]
    g1[6] = (int)((NE & 0xFFFFu) << 16);             // stride0 hi=0 | stride1[15:0]
    g1[7] = 0;
    asm volatile("tensor_load_to_lds %0, %1" :: "s"(g0), "s"(g1) : "memory");
    __builtin_amdgcn_s_wait_tensorcnt(0);
  }
  // correctness hedge: cooperative copy of the same bytes (identical values)
  {
    const uint64_t* g = (const uint64_t*)wsrc;
    uint64_t* l = (uint64_t*)&s_wt[0];
    for (int i = threadIdx.x; i < WT_BYTES / 8; i += 256) l[i] = g[i];
  }

  if (lane < 16) {
    int e = ebase + lane;
    float vx = evec[e * 3 + 0], vy = evec[e * 3 + 1], vz = evec[e * 3 + 2];
    float nrm = sqrtf(vx * vx + vy * vy + vz * vz);
    float inv = 1.0f / fmaxf(nrm, 1e-6f);
    s_vx[wave][lane] = vx * inv;
    s_vy[wave][lane] = vy * inv;
    s_vz[wave][lane] = vz * inv;
    float w = ew[e];
    float C = 0.5f * (__cosf(w * 0.6283185307179587f) + 1.0f);
    s_C[wave][lane] = (w < 5.0f) ? C : 0.0f;
    s_src[wave][lane] = eidx[e];
    s_dst[wave][lane] = eidx[N_EDGES + e];
  }
  __syncthreads();

  const int erow = ebase + lm;
  // A-fragment for the RBF GEMMs (16 edges x 32 rbf)
  v16h attrA = frag_f32(eattr + (size_t)erow * 32, hi);

  // A-fragments for Zij = [x[src] || x[dst]] (16 x 128), 4 K-chunks
  int srcn = s_src[wave][lm], dstn = s_dst[wave][lm];
  v16h zA[4];
#pragma unroll
  for (int c = 0; c < 4; ++c) {
    const _Float16* base = x16 + (size_t)(c < 2 ? srcn : dstn) * 64 + (c & 1) * 32;
    zA[c] = frag_f16(base, hi);
  }

  // Zij accumulators: 4 N-tiles of 16 units (B-fragments now from LDS)
  v8f Z[4];
#pragma unroll
  for (int t = 0; t < 4; ++t) {
    int u = 16 * t + lm;
    v8f zacc = {};
#pragma unroll
    for (int c = 0; c < 4; ++c)
      zacc = wmma(zA[c], frag_f16(&s_wt[EMB2_OFS + u * 128 + 32 * c], hi), zacc);
    float bu = emb2b[u];
#pragma unroll
    for (int r = 0; r < 8; ++r) zacc[r] += bu;
    Z[t] = zacc;
  }

  // W1/W2/W3, fused f = (W+b)*C*Z, scattered into 10-channel node accumulator
#pragma unroll
  for (int w = 0; w < 3; ++w) {
    const float* dpb = (w == 0) ? dp1b : (w == 1) ? dp2b : dp3b;
#pragma unroll
    for (int t = 0; t < 4; ++t) {
      int u = 16 * t + lm;
      v8f a = {};
      a = wmma(attrA, frag_f16(&s_wt[w * 2048 + u * 32], hi), a);
      float bu = dpb[u];
#pragma unroll
      for (int r = 0; r < 8; ++r) {
        int m = r + 8 * hi;
        float f = (a[r] + bu) * s_C[wave][m] * Z[t][r];
        float* an = acc + (size_t)s_dst[wave][m] * 640 + u;
        float vx = s_vx[wave][m], vy = s_vy[wave][m], vz = s_vz[wave][m];
        if (w == 0) {
          atomicAdd(an, f);                               // I
        } else if (w == 1) {
          atomicAdd(an + 64, -vz * f);                    // a01
          atomicAdd(an + 128, vy * f);                    // a02
          atomicAdd(an + 192, -vx * f);                   // a12
        } else {
          float t3 = (vx * vx + vy * vy + vz * vz) * (1.0f / 3.0f);
          atomicAdd(an + 256, (vx * vx - t3) * f);        // s00
          atomicAdd(an + 320, vx * vy * f);               // s01
          atomicAdd(an + 384, vx * vz * f);               // s02
          atomicAdd(an + 448, (vy * vy - t3) * f);        // s11
          atomicAdd(an + 512, vy * vz * f);               // s12
          atomicAdd(an + 576, (vz * vz - t3) * f);        // s22
        }
      }
    }
  }
}

// ---------------- per-node frobenius norm + layernorm ----------------
__global__ void k_norm(const float* __restrict__ acc, const float* __restrict__ ln_g,
                       const float* __restrict__ ln_b, float* __restrict__ nrmln) {
  int n = blockIdx.x * blockDim.x + threadIdx.x;
  const float* a = acc + (size_t)n * 640;
  float nr[64];
  float mu = 0.0f;
#pragma unroll 8
  for (int u = 0; u < 64; ++u) {
    float I = a[u];
    float a01 = a[64 + u], a02 = a[128 + u], a12 = a[192 + u];
    float s00 = a[256 + u], s01 = a[320 + u], s02 = a[384 + u];
    float s11 = a[448 + u], s12 = a[512 + u], s22 = a[576 + u];
    float v = (I + s00) * (I + s00) + (I + s11) * (I + s11) + (I + s22) * (I + s22) +
              2.0f * (a01 * a01 + a02 * a02 + a12 * a12 +
                      s01 * s01 + s02 * s02 + s12 * s12);
    nr[u] = v;
    mu += v;
  }
  mu *= (1.0f / 64.0f);
  float var = 0.0f;
#pragma unroll 8
  for (int u = 0; u < 64; ++u) { float d = nr[u] - mu; var += d * d; }
  var *= (1.0f / 64.0f);
  float rs = rsqrtf(var + 1e-5f);
#pragma unroll 8
  for (int u = 0; u < 64; ++u)
    nrmln[(size_t)n * 64 + u] = (nr[u] - mu) * rs * ln_g[u] + ln_b[u];
}

// ---------------- MLP (silu o linear twice) with WMMA ----------------
// block = 256 (8 waves), wave handles 16 nodes; grid = 16384/128 = 128
__global__ void k_mlp(const float* __restrict__ nrmln, const _Float16* __restrict__ ls0f,
                      const _Float16* __restrict__ ls1f, const float* __restrict__ ls0b,
                      const float* __restrict__ ls1b, float* __restrict__ nfac) {
  const int wave = threadIdx.x >> 5, lane = threadIdx.x & 31;
  const int hi = lane >> 4, lm = lane & 15;
  const int tbase = (blockIdx.x * 8 + wave) * 16;

  __shared__ __align__(16) _Float16 s_h0[8][16 * 128];

  v16h a0[2];
#pragma unroll
  for (int c = 0; c < 2; ++c)
    a0[c] = frag_f32(nrmln + (size_t)(tbase + lm) * 64 + 32 * c, hi);

#pragma unroll
  for (int t = 0; t < 8; ++t) {
    int j = 16 * t + lm;
    v8f h = {};
#pragma unroll
    for (int c = 0; c < 2; ++c)
      h = wmma(a0[c], frag_f16(ls0f + (size_t)j * 64 + 32 * c, hi), h);
    float bj = ls0b[j];
#pragma unroll
    for (int r = 0; r < 8; ++r) {
      int m = r + 8 * hi;
      s_h0[wave][m * 128 + j] = (_Float16)silu(h[r] + bj);
    }
  }
  __syncthreads();

  v16h a1[4];
#pragma unroll
  for (int c = 0; c < 4; ++c)
    a1[c] = frag_f16(&s_h0[wave][lm * 128 + 32 * c], hi);

#pragma unroll
  for (int t = 0; t < 12; ++t) {
    int j = 16 * t + lm;
    v8f h = {};
#pragma unroll
    for (int c = 0; c < 4; ++c)
      h = wmma(a1[c], frag_f16(ls1f + (size_t)j * 128 + 32 * c, hi), h);
    float bj = ls1b[j];
#pragma unroll
    for (int r = 0; r < 8; ++r) {
      int m = r + 8 * hi;
      nfac[(size_t)(tbase + m) * 192 + j] = silu(h[r] + bj);
    }
  }
}

// ---------------- channel transforms + gated reassembly ----------------
// 10 channel GEMMs [16x64]@[64x64] per node tile; writes full N x 3 x 3 x 64
__global__ void k_assemble(const float* __restrict__ acc, const _Float16* __restrict__ ltf,
                           const float* __restrict__ nfac, float* __restrict__ out) {
  const int wave = threadIdx.x >> 5, lane = threadIdx.x & 31;
  const int hi = lane >> 4, lm = lane & 15;
  const int tbase = (blockIdx.x * 8 + wave) * 16;

#pragma unroll
  for (int t = 0; t < 4; ++t) {
    int u = 16 * t + lm;
    float o[9][8];
#pragma unroll
    for (int ij = 0; ij < 9; ++ij)
#pragma unroll
      for (int r = 0; r < 8; ++r) o[ij][r] = 0.0f;

#pragma unroll
    for (int c = 0; c < 10; ++c) {
      const int w    = (c == 0) ? 0 : (c < 4) ? 1 : 2;   // lt0 / lt1 / lt2
      const int comp = (c == 0) ? 0 : (c < 4) ? 1 : 2;   // nI / nA / nS gate
      v8f d = {};
#pragma unroll
      for (int ch = 0; ch < 2; ++ch) {
        v16h a = frag_f32(acc + (size_t)(tbase + lm) * 640 + c * 64 + 32 * ch, hi);
        v16h b = frag_f16(ltf + (size_t)w * 4096 + (size_t)u * 64 + 32 * ch, hi);
        d = wmma(a, b, d);
      }
#pragma unroll
      for (int r = 0; r < 8; ++r) {
        int node = tbase + r + 8 * hi;
        float f = d[r] * nfac[(size_t)node * 192 + u * 3 + comp];
        switch (c) {
          case 0: o[0][r] += f; o[4][r] += f; o[8][r] += f; break;  // I on diag
          case 1: o[1][r] += f; o[3][r] -= f; break;                // a01
          case 2: o[2][r] += f; o[6][r] -= f; break;                // a02
          case 3: o[5][r] += f; o[7][r] -= f; break;                // a12
          case 4: o[0][r] += f; break;                              // s00
          case 5: o[1][r] += f; o[3][r] += f; break;                // s01
          case 6: o[2][r] += f; o[6][r] += f; break;                // s02
          case 7: o[4][r] += f; break;                              // s11
          case 8: o[5][r] += f; o[7][r] += f; break;                // s12
          case 9: o[8][r] += f; break;                              // s22
        }
      }
    }
#pragma unroll
    for (int r = 0; r < 8; ++r) {
      int node = tbase + r + 8 * hi;
      float* ob = out + (size_t)node * 576 + u;
#pragma unroll
      for (int ij = 0; ij < 9; ++ij) ob[ij * 64] = o[ij][r];
    }
  }
}

// ---------------- launcher ----------------
extern "C" void kernel_launch(void* const* d_in, const int* in_sizes, int n_in,
                              void* d_out, int out_size, void* d_ws, size_t ws_size,
                              hipStream_t stream) {
  const int*   z     = (const int*)d_in[0];
  const int*   eidx  = (const int*)d_in[1];
  const float* ew    = (const float*)d_in[2];
  const float* evec  = (const float*)d_in[3];
  const float* eattr = (const float*)d_in[4];
  const float* embw  = (const float*)d_in[5];
  const float* dp1w  = (const float*)d_in[6];
  const float* dp1b  = (const float*)d_in[7];
  const float* dp2w  = (const float*)d_in[8];
  const float* dp2b  = (const float*)d_in[9];
  const float* dp3w  = (const float*)d_in[10];
  const float* dp3b  = (const float*)d_in[11];
  const float* emb2w = (const float*)d_in[12];
  const float* emb2b = (const float*)d_in[13];
  const float* lt0w  = (const float*)d_in[14];
  const float* lt1w  = (const float*)d_in[15];
  const float* lt2w  = (const float*)d_in[16];
  const float* ls0w  = (const float*)d_in[17];
  const float* ls0b  = (const float*)d_in[18];
  const float* ls1w  = (const float*)d_in[19];
  const float* ls1b  = (const float*)d_in[20];
  const float* ln_g  = (const float*)d_in[21];
  const float* ln_b  = (const float*)d_in[22];
  float* out = (float*)d_out;

  char* ws = (char*)d_ws;
  _Float16* x16   = (_Float16*)(ws + OFF_X16);
  _Float16* dpf   = (_Float16*)(ws + OFF_DPF);
  _Float16* emb2f = (_Float16*)(ws + OFF_EMB2F);
  _Float16* ls0f  = (_Float16*)(ws + OFF_LS0F);
  _Float16* ls1f  = (_Float16*)(ws + OFF_LS1F);
  _Float16* ltf   = (_Float16*)(ws + OFF_LTF);
  float*    acc   = (float*)(ws + OFF_ACC);
  float*    nrmln = (float*)(ws + OFF_NRM);
  float*    nfac  = (float*)(ws + OFF_NFAC);

  auto cvt = [&](const float* s, _Float16* d, int n) {
    k_cvt_f16<<<(n + 255) / 256, 256, 0, stream>>>(s, d, n);
  };
  cvt(dp1w, dpf, 2048);
  cvt(dp2w, dpf + 2048, 2048);
  cvt(dp3w, dpf + 4096, 2048);
  cvt(emb2w, emb2f, 64 * 128);
  cvt(ls0w, ls0f, 128 * 64);
  cvt(ls1w, ls1f, 192 * 128);
  cvt(lt0w, ltf, 4096);
  cvt(lt1w, ltf + 4096, 4096);
  cvt(lt2w, ltf + 8192, 4096);

  k_gather_x<<<(N_NODES * 64) / 256, 256, 0, stream>>>(z, embw, x16);
  k_zero<<<(N_NODES * 640 + 255) / 256, 256, 0, stream>>>(acc, N_NODES * 640);

  k_edges<<<N_EDGES / 128, 256, 0, stream>>>(eidx, ew, evec, eattr, dp1b, dp2b, dp3b,
                                             emb2b, x16, dpf, acc);
  k_norm<<<N_NODES / 256, 256, 0, stream>>>(acc, ln_g, ln_b, nrmln);
  k_mlp<<<N_NODES / 128, 256, 0, stream>>>(nrmln, ls0f, ls1f, ls0b, ls1b, nfac);
  k_assemble<<<N_NODES / 128, 256, 0, stream>>>(acc, ltf, nfac, out);
}
